// ESMMimicryModule_38285338477246
// MI455X (gfx1250) — compile-verified, matrix operand order checked by reading, not run on hardware
//
#include <hip/hip_runtime.h>
#include <hip/hip_bf16.h>

// ---------------------------------------------------------------------------
// Shapes
// ---------------------------------------------------------------------------
#define B_    8
#define S_    2048
#define H_    8
#define NL_   128
#define DF_   64
#define VF_   64
#define NF_   1280
#define HS_   33
#define FEAT_ 1313          // NF + HS
#define KPAD_ 1344          // FEAT+1 padded up to multiple of 64
#define BH_   64            // B*H
#define QKSCALE 0.594603557501360533f   // 1 / 64^0.125

typedef _Float16 f16_t;
typedef __attribute__((ext_vector_type(16))) _Float16 v16h;
typedef __attribute__((ext_vector_type(8)))  float    v8f;
typedef __attribute__((ext_vector_type(4)))  unsigned int u32x4;
typedef __attribute__((ext_vector_type(8)))  int      i32x8;
typedef __attribute__((ext_vector_type(4)))  int      i32x4;

union FragH { v16h v; unsigned u[8]; f16_t h[16]; };

#ifdef __has_builtin
#  if __has_builtin(__builtin_amdgcn_tensor_load_to_lds) && \
      __has_builtin(__builtin_amdgcn_s_wait_tensorcnt)
#    define HAVE_TDM 1
#  endif
#endif
#ifndef HAVE_TDM
#  define HAVE_TDM 0
#endif

// ---------------------------------------------------------------------------
// Generic batched GEMM: C[MxN] = A[MxK](f16,row-major) * B[KxN](f16,row-major)
// fp32 accumulate via v_wmma_f32_16x16x32_f16.  Block = 8 waves, each wave
// computes a 32x32 tile; block tile = 128(M) x 64(N).  M%128==0, N%64==0,
// K%32==0 guaranteed by buffer padding.
// ---------------------------------------------------------------------------
template<bool OUTF16>
__global__ __launch_bounds__(256) void k_gemm(
    const f16_t* __restrict__ A, const f16_t* __restrict__ Bm,
    float* __restrict__ C32, f16_t* __restrict__ C16,
    int M, int K, int lda, int ldb, int ldc,
    long aStride, int aDiv, long bStride, int bDiv, long cStride)
{
  const int z = blockIdx.z;
  A  += (size_t)(z / aDiv) * aStride;
  Bm += (size_t)(z / bDiv) * bStride;
  const size_t cbase = (size_t)z * cStride;

  const int wave = threadIdx.x >> 5;
  const int lane = threadIdx.x & 31;
  const int m0 = blockIdx.y * 128 + (wave >> 1) * 32;
  const int n0 = blockIdx.x * 64  + (wave & 1)  * 32;
  const int r  = lane & 15;
  const int hi = lane >> 4;

  v8f z8 = {0.f,0.f,0.f,0.f,0.f,0.f,0.f,0.f};
  v8f acc[2][2];
  acc[0][0] = z8; acc[0][1] = z8; acc[1][0] = z8; acc[1][1] = z8;

  for (int k0 = 0; k0 < K; k0 += 32) {
    FragH a[2], b[2];
#pragma unroll
    for (int i = 0; i < 2; ++i) {
      const f16_t* ap = A + (size_t)(m0 + i*16 + r) * lda + k0;
#pragma unroll
      for (int v = 0; v < 8; ++v) {
        const int kk = ((v & 3) << 1) + (hi << 3) + ((v >> 2) << 4);
        a[i].u[v] = *(const unsigned*)(const void*)(ap + kk);
      }
      if (k0 + 32 < K) __builtin_prefetch(ap + 32, 0, 3);
    }
#pragma unroll
    for (int j = 0; j < 2; ++j) {
      const int n = n0 + j*16 + r;
#pragma unroll
      for (int v = 0; v < 8; ++v) {
        const int kk = k0 + ((v & 3) << 1) + (hi << 3) + ((v >> 2) << 4);
        b[j].h[2*v  ] = Bm[(size_t)kk      * ldb + n];
        b[j].h[2*v+1] = Bm[(size_t)(kk+1) * ldb + n];
      }
    }
#pragma unroll
    for (int i = 0; i < 2; ++i)
#pragma unroll
      for (int j = 0; j < 2; ++j)
        acc[i][j] = __builtin_amdgcn_wmma_f32_16x16x32_f16(
            false, a[i].v, false, b[j].v, (short)0, acc[i][j], false, false);
  }

#pragma unroll
  for (int i = 0; i < 2; ++i)
#pragma unroll
    for (int j = 0; j < 2; ++j) {
      const int rr = m0 + i*16 + hi*8;
      const int cc = n0 + j*16 + r;
#pragma unroll
      for (int q = 0; q < 8; ++q) {
        const size_t idx = cbase + (size_t)(rr + q) * ldc + cc;
        if (OUTF16) C16[idx] = (f16_t)acc[i][j][q];
        else        C32[idx] = acc[i][j][q];
      }
    }
}

static inline void gemm(hipStream_t st, const void* A, const void* Bm, void* C,
                        bool f16out, int M, int N, int K, int lda, int ldb, int ldc,
                        long aS, int aD, long bS, int bD, long cS, int batch)
{
  dim3 g(N / 64, M / 128, batch), blk(256);
  if (f16out)
    k_gemm<true ><<<g, blk, 0, st>>>((const f16_t*)A, (const f16_t*)Bm,
        (float*)nullptr, (f16_t*)C, M, K, lda, ldb, ldc, aS, aD, bS, bD, cS);
  else
    k_gemm<false><<<g, blk, 0, st>>>((const f16_t*)A, (const f16_t*)Bm,
        (float*)C, (f16_t*)nullptr, M, K, lda, ldb, ldc, aS, aD, bS, bD, cS);
}

// ---------------------------------------------------------------------------
// Weight-stationary GEMM with Tensor Data Mover: one wave issues an async
// 2-D tile DMA (32 K-rows x 64 N-cols, f16) of B into LDS per K-step,
// double-buffered and overlapped with WMMA compute; all 8 waves gather B
// fragments from LDS.  D# built per CDNA5 ISA.
// This toolchain's builtin carries 6 operands (g0, g1, g2, g3, g4, cpol).
// ---------------------------------------------------------------------------
#if HAVE_TDM
__device__ __forceinline__ void tdm_load_b_tile(const f16_t* gsrc, int ldbElems,
                                                void* ldsDst)
{
  const unsigned long long gaddr = (unsigned long long)(uintptr_t)gsrc;
  const unsigned laddr = (unsigned)(uintptr_t)ldsDst;   // low 32 bits = LDS offset
  u32x4 g0;
  g0.x = 1u;                                            // count=1 valid descriptor
  g0.y = laddr;                                         // lds_addr (bytes)
  g0.z = (unsigned)gaddr;                               // global_addr[31:0]
  g0.w = ((unsigned)(gaddr >> 32) & 0x01FFFFFFu) | 0x80000000u; // addr[56:32] | type=2
  const unsigned td0 = (unsigned)ldbElems;              // tensor_dim0 (elements)
  const unsigned td1 = 32u;                             // tensor_dim1 (rows)
  i32x8 g1;
  g1.s0 = (int)(1u << 16);                              // data_size=1 (2B); wg_mask=0
  g1.s1 = (int)((td0 & 0xFFFFu) << 16);                 // atomic_barrier=0 | td0[15:0]
  g1.s2 = (int)(((td0 >> 16) & 0xFFFFu) | ((td1 & 0xFFFFu) << 16));
  g1.s3 = (int)(((td1 >> 16) & 0xFFFFu) | (64u << 16)); // tile_dim0 = 64
  g1.s4 = (int)32u;                                     // tile_dim1 = 32, tile_dim2 = 0
  g1.s5 = (int)td0;                                     // tensor_dim0_stride[31:0]
  g1.s6 = 0;                                            // stride hi | dim1_stride lo
  g1.s7 = 0;
  i32x4 zz  = {0, 0, 0, 0};
  i32x8 zz8 = {0, 0, 0, 0, 0, 0, 0, 0};
  __builtin_amdgcn_tensor_load_to_lds(g0, g1, zz, zz, zz8, 0);
}
#endif

template<bool OUTF16>
__global__ __launch_bounds__(256) void k_gemm_tdm(
    const f16_t* __restrict__ A, const f16_t* __restrict__ Bm,
    float* __restrict__ C32, f16_t* __restrict__ C16,
    int M, int K, int lda, int ldb, int ldc)
{
  __shared__ __align__(16) f16_t tile[2][32 * 64];

  const int wave = threadIdx.x >> 5;
  const int lane = threadIdx.x & 31;
  const int m0  = blockIdx.y * 128 + (wave >> 1) * 32;
  const int nb  = blockIdx.x * 64;          // block N origin (global)
  const int n0w = (wave & 1) * 32;          // wave N origin inside tile
  const int r   = lane & 15;
  const int hi  = lane >> 4;

  v8f z8 = {0.f,0.f,0.f,0.f,0.f,0.f,0.f,0.f};
  v8f acc[2][2];
  acc[0][0] = z8; acc[0][1] = z8; acc[1][0] = z8; acc[1][1] = z8;

  auto issue = [&](int buf, int k0) {
#if HAVE_TDM
    if (threadIdx.x < 32)
      tdm_load_b_tile(Bm + (size_t)k0 * ldb + nb, ldb, &tile[buf][0]);
#else
    for (int idx = threadIdx.x; idx < 32 * 32; idx += 256) {   // u32 copies
      const int row = idx >> 5, cp = idx & 31;
      *(unsigned*)&tile[buf][row * 64 + cp * 2] =
          *(const unsigned*)(const void*)&Bm[(size_t)(k0 + row) * ldb + nb + cp * 2];
    }
#endif
  };
  auto settle = [&]() {
#if HAVE_TDM
    if (threadIdx.x < 32) __builtin_amdgcn_s_wait_tensorcnt(0);
#endif
    __syncthreads();
  };

  issue(0, 0);
  settle();

  for (int k0 = 0; k0 < K; k0 += 32) {
    const int cur = (k0 >> 5) & 1;
    if (k0 + 32 < K) issue(cur ^ 1, k0 + 32);   // overlap DMA with compute

    FragH a[2], b[2];
#pragma unroll
    for (int i = 0; i < 2; ++i) {
      const f16_t* ap = A + (size_t)(m0 + i*16 + r) * lda + k0;
#pragma unroll
      for (int v = 0; v < 8; ++v) {
        const int kk = ((v & 3) << 1) + (hi << 3) + ((v >> 2) << 4);
        a[i].u[v] = *(const unsigned*)(const void*)(ap + kk);
      }
      if (k0 + 32 < K) __builtin_prefetch(ap + 32, 0, 3);
    }
    const f16_t* bt = &tile[cur][0];
#pragma unroll
    for (int j = 0; j < 2; ++j) {
      const int n = n0w + j*16 + r;
#pragma unroll
      for (int v = 0; v < 8; ++v) {
        const int kk = ((v & 3) << 1) + (hi << 3) + ((v >> 2) << 4);
        b[j].h[2*v  ] = bt[kk       * 64 + n];
        b[j].h[2*v+1] = bt[(kk + 1) * 64 + n];
      }
    }
#pragma unroll
    for (int i = 0; i < 2; ++i)
#pragma unroll
      for (int j = 0; j < 2; ++j)
        acc[i][j] = __builtin_amdgcn_wmma_f32_16x16x32_f16(
            false, a[i].v, false, b[j].v, (short)0, acc[i][j], false, false);

    settle();   // next tile resident before it becomes 'cur'
  }

#pragma unroll
  for (int i = 0; i < 2; ++i)
#pragma unroll
    for (int j = 0; j < 2; ++j) {
      const int rr = m0 + i*16 + hi*8;
      const int cc = nb + n0w + j*16 + r;
#pragma unroll
      for (int q = 0; q < 8; ++q) {
        const size_t idx = (size_t)(rr + q) * ldc + cc;
        if (OUTF16) C16[idx] = (f16_t)acc[i][j][q];
        else        C32[idx] = acc[i][j][q];
      }
    }
}

static inline void gemm_tdm(hipStream_t st, const void* A, const void* Bm, void* C,
                            bool f16out, int M, int N, int K, int lda, int ldb, int ldc)
{
  dim3 g(N / 64, M / 128, 1), blk(256);
  if (f16out)
    k_gemm_tdm<true ><<<g, blk, 0, st>>>((const f16_t*)A, (const f16_t*)Bm,
        (float*)nullptr, (f16_t*)C, M, K, lda, ldb, ldc);
  else
    k_gemm_tdm<false><<<g, blk, 0, st>>>((const f16_t*)A, (const f16_t*)Bm,
        (float*)C, (f16_t*)nullptr, M, K, lda, ldb, ldc);
}

// ---------------------------------------------------------------------------
// orig = concat(x_embeds, softmax(latent)*mask), +constant-1 feature, f16 pad
// ---------------------------------------------------------------------------
__global__ void k_build_orig(const float* __restrict__ latent,
                             const float* __restrict__ masks,
                             const float* __restrict__ xemb,
                             f16_t* __restrict__ orig)
{
  const int bs = blockIdx.x;
  __shared__ float sm[HS_];
  __shared__ float ssum;
  const int t = threadIdx.x;
  if (t < HS_) sm[t] = latent[(size_t)bs * HS_ + t];
  __syncthreads();
  if (t == 0) {
    float mx = -1e30f;
    for (int i = 0; i < HS_; ++i) mx = fmaxf(mx, sm[i]);
    float s = 0.f;
    for (int i = 0; i < HS_; ++i) { sm[i] = __expf(sm[i] - mx); s += sm[i]; }
    ssum = s;
  }
  __syncthreads();
  const float inv = 1.f / ssum;
  const float mk  = masks[bs];
  for (int f = t; f < KPAD_; f += blockDim.x) {
    float v;
    if      (f < NF_)    v = xemb[(size_t)bs * NF_ + f];
    else if (f < FEAT_)  v = sm[f - NF_] * inv * mk;
    else if (f == FEAT_) v = 1.f;
    else                 v = 0.f;
    orig[(size_t)bs * KPAD_ + f] = (f16_t)v;
  }
}

// ---------------------------------------------------------------------------
// Fused per-layer weight matrix (KPAD x Npad), bias row duplicated at k=1313,
// Q/K columns pre-scaled by 1/DF^0.125.
// ---------------------------------------------------------------------------
__global__ void k_prep_w(const float* __restrict__ q, const float* __restrict__ k,
                         const float* __restrict__ v, const float* __restrict__ p,
                         f16_t* __restrict__ W, int L, int Npad)
{
  const int krow = blockIdx.x;
  const int ks = (krow < FEAT_) ? krow : (krow == FEAT_ ? FEAT_ - 1 : -1);
  for (int col = threadIdx.x; col < Npad; col += blockDim.x) {
    float val = 0.f;
    if (ks >= 0) {
      if (col < 512) {
        const int h = col >> 6, d = col & 63;
        val = q[(((size_t)L * H_ + h) * (FEAT_ + 1) + ks) * DF_ + d] * QKSCALE;
      } else if (col < 576) {
        val = k[((size_t)L * (FEAT_ + 1) + ks) * DF_ + (col - 512)] * QKSCALE;
      } else if (col < 1088) {
        const int h = (col - 576) >> 6, d = (col - 576) & 63;
        val = v[(((size_t)L * H_ + h) * (FEAT_ + 1) + ks) * VF_ + d];
      } else if (L == 0 && col < 1152) {
        val = p[((size_t)0 * (FEAT_ + 1) + ks) * VF_ + (col - 1088)];
      } else if (L == 0 && col < 1216) {
        val = p[((size_t)1 * (FEAT_ + 1) + ks) * VF_ + (col - 1152)];
      }
    }
    W[(size_t)krow * Npad + col] = (f16_t)val;
  }
}

__global__ void k_prep_collect(const float* __restrict__ cf,
                               const float* __restrict__ cc,
                               f16_t* __restrict__ c0, f16_t* __restrict__ c1)
{
  const int kr = blockIdx.x; // 0..511
  for (int col = threadIdx.x; col < KPAD_; col += blockDim.x)
    c0[(size_t)kr * KPAD_ + col] = (f16_t)(col < FEAT_ ? cf[(size_t)kr * FEAT_ + col] : 0.f);
  for (int col = threadIdx.x; col < 64; col += blockDim.x)
    c1[(size_t)kr * 64 + col]   = (f16_t)(col < HS_  ? cc[(size_t)kr * HS_  + col] : 0.f);
}

// ---------------------------------------------------------------------------
// Landmark means from E (fp32, ld = Npad)
// ---------------------------------------------------------------------------
__global__ void k_landmark(const float* __restrict__ E, int Npad,
                           f16_t* __restrict__ qland, f16_t* __restrict__ klandt)
{
  const int b = blockIdx.x >> 7, l = blockIdx.x & 127, d = threadIdx.x;
  for (int h = 0; h < H_; ++h) {
    float s = 0.f;
    for (int t = 0; t < 16; ++t)
      s += E[((size_t)b * S_ + l * 16 + t) * Npad + h * 64 + d];
    qland[(((size_t)b * H_ + h) * NL_ + l) * DF_ + d] = (f16_t)(s * (1.f / 16.f));
  }
  float s = 0.f;
  for (int t = 0; t < 16; ++t)
    s += E[((size_t)b * S_ + l * 16 + t) * Npad + 512 + d];
  klandt[((size_t)b * DF_ + d) * NL_ + l] = (f16_t)(s * (1.f / 16.f));
}

__global__ void k_extract(const float* __restrict__ E, int Npad, int hasProj,
                          f16_t* __restrict__ eq, f16_t* __restrict__ ekt,
                          f16_t* __restrict__ ev,
                          float* __restrict__ proj0, float* __restrict__ proj1)
{
  const int bs = blockIdx.x, b = bs >> 11, s = bs & 2047;
  for (int col = threadIdx.x; col < 1088; col += blockDim.x) {
    const float val = E[(size_t)bs * Npad + col];
    if (col < 512)
      eq[(((size_t)b * H_ + (col >> 6)) * S_ + s) * DF_ + (col & 63)] = (f16_t)val;
    else if (col < 576)
      ekt[((size_t)b * DF_ + (col - 512)) * S_ + s] = (f16_t)val;
    else
      ev[(((size_t)b * H_ + ((col - 576) >> 6)) * S_ + s) * VF_ + ((col - 576) & 63)] = (f16_t)val;
  }
  if (hasProj && threadIdx.x < 128) {
    const int col = 1088 + threadIdx.x;
    const float val = E[(size_t)bs * Npad + col];
    if (col < 1152) proj0[(size_t)bs * VF_ + (col - 1088)] = val;
    else            proj1[(size_t)bs * VF_ + (col - 1152)] = val;
  }
}

__global__ void k_possel(const float* __restrict__ agg, int L,
                         float* __restrict__ ps)
{
  const int h = blockIdx.x >> 7, l = blockIdx.x & 127;
  const float sig = fabsf(agg[L * H_ + h]) + 0.001f;
  const int base = l * 16;
  for (int j = threadIdx.x; j < S_; j += blockDim.x) {
    float s = 0.f;
    for (int t = 0; t < 16; ++t) s += fabsf((float)(base + t - j));
    ps[((size_t)h * NL_ + l) * S_ + j] = -sig * s * (1.f / 16.f);
  }
}

__global__ void k_possel2(const float* __restrict__ ps, float* __restrict__ ps2)
{
  const int h = blockIdx.x >> 7, l = blockIdx.x & 127, m = threadIdx.x;
  float s = 0.f;
  for (int t = 0; t < 16; ++t) s += ps[((size_t)h * NL_ + l) * S_ + m * 16 + t];
  ps2[((size_t)h * NL_ + l) * NL_ + m] = s * (1.f / 16.f);
}

__global__ __launch_bounds__(128) void k_softmax1(
    const float* __restrict__ pre, const float* __restrict__ ps,
    f16_t* __restrict__ outh)
{
  const int row = blockIdx.x;                  // (b*H+h)*S + s
  const int h = (row >> 11) & 7, s = row & 2047, l = threadIdx.x;
  float val = pre[(size_t)row * NL_ + l] + ps[((size_t)h * NL_ + l) * S_ + s];
  __shared__ float red[128];
  red[l] = val; __syncthreads();
  for (int st = 64; st > 0; st >>= 1) { if (l < st) red[l] = fmaxf(red[l], red[l + st]); __syncthreads(); }
  const float mx = red[0]; __syncthreads();
  const float e = __expf(val - mx);
  red[l] = e; __syncthreads();
  for (int st = 64; st > 0; st >>= 1) { if (l < st) red[l] += red[l + st]; __syncthreads(); }
  outh[(size_t)row * NL_ + l] = (f16_t)(e / red[0]);
}

__global__ __launch_bounds__(128) void k_softmax2(
    const float* __restrict__ pre, const float* __restrict__ ps2,
    float* __restrict__ key32, f16_t* __restrict__ keyh)
{
  const int z = blockIdx.x, h = z & 7, i = threadIdx.x;
  const size_t base = (size_t)z * NL_ * NL_ + (size_t)i * NL_;
  const size_t pbase = ((size_t)h * NL_ + i) * NL_;
  float v[128];
  float mx = -1e30f;
  for (int j = 0; j < NL_; ++j) { v[j] = pre[base + j] + ps2[pbase + j]; mx = fmaxf(mx, v[j]); }
  float s = 0.f;
  for (int j = 0; j < NL_; ++j) { v[j] = __expf(v[j] - mx); s += v[j]; }
  const float inv = 1.f / s;
  for (int j = 0; j < NL_; ++j) {
    const float rr = v[j] * inv;
    key32[base + j] = rr;
    keyh[base + j]  = (f16_t)rr;
  }
}

__global__ __launch_bounds__(128) void k_colmax(const float* __restrict__ key32,
                                                float* __restrict__ gmax)
{
  const int z = blockIdx.x, j = threadIdx.x;
  float cs = 0.f;
  for (int i = 0; i < NL_; ++i) cs += key32[(size_t)z * NL_ * NL_ + (size_t)i * NL_ + j];
  __shared__ float red[128];
  red[j] = cs; __syncthreads();
  for (int st = 64; st > 0; st >>= 1) { if (j < st) red[j] = fmaxf(red[j], red[j + st]); __syncthreads(); }
  if (j == 0) atomicMax((unsigned int*)gmax, __float_as_uint(red[0]));
}

__global__ void k_zero1(float* p) { p[0] = 0.f; }

__global__ __launch_bounds__(128) void k_initval(const float* __restrict__ key32,
                                                 const float* __restrict__ gmax,
                                                 f16_t* __restrict__ valh)
{
  const int z = blockIdx.x >> 7, i = blockIdx.x & 127, j = threadIdx.x;
  const float s = 1.f / gmax[0];
  valh[(size_t)z * NL_ * NL_ + (size_t)i * NL_ + j] =
      (f16_t)(key32[(size_t)z * NL_ * NL_ + (size_t)j * NL_ + i] * s);
}

__global__ __launch_bounds__(128) void k_polyh(const float* __restrict__ X,
                                               f16_t* __restrict__ outH,
                                               f16_t* __restrict__ copyH,
                                               float alpha, float beta)
{
  const size_t idx = (size_t)blockIdx.x * 128 + threadIdx.x;
  const int i = blockIdx.x & 127, j = threadIdx.x;
  const float x = X[idx];
  outH[idx] = (f16_t)(alpha * (i == j ? 1.f : 0.f) + beta * x);
  if (copyH) copyH[idx] = (f16_t)x;
}

__global__ __launch_bounds__(256) void k_softmax3(
    const float* __restrict__ scores, const float* __restrict__ ps,
    const float* __restrict__ masks, f16_t* __restrict__ outh)
{
  const int row = blockIdx.x;                 // (b*H+h)*NL + l
  const int b = row >> 10, h = (row >> 7) & 7, l = row & 127;
  const size_t sbase = (size_t)row * S_;
  const size_t pbase = ((size_t)h * NL_ + l) * S_;
  float v[8]; float mx = -1e30f;
  for (int k = 0; k < 8; ++k) {
    const int j = threadIdx.x + k * 256;
    float t = scores[sbase + j] + ps[pbase + j];
    if (masks[(size_t)b * S_ + j] == 0.f) t += -1000000.0f;
    v[k] = t; mx = fmaxf(mx, t);
  }
  __shared__ float red[256];
  red[threadIdx.x] = mx; __syncthreads();
  for (int st = 128; st > 0; st >>= 1) { if (threadIdx.x < st) red[threadIdx.x] = fmaxf(red[threadIdx.x], red[threadIdx.x + st]); __syncthreads(); }
  mx = red[0]; __syncthreads();
  float s = 0.f;
  for (int k = 0; k < 8; ++k) { v[k] = __expf(v[k] - mx); s += v[k]; }
  red[threadIdx.x] = s; __syncthreads();
  for (int st = 128; st > 0; st >>= 1) { if (threadIdx.x < st) red[threadIdx.x] += red[threadIdx.x + st]; __syncthreads(); }
  const float inv = 1.f / red[0];
  for (int k = 0; k < 8; ++k)
    outh[sbase + threadIdx.x + k * 256] = (f16_t)(v[k] * inv);
}

__global__ void k_f32tof16(const float* __restrict__ s, f16_t* __restrict__ d, long n)
{
  const long i = (long)blockIdx.x * blockDim.x + threadIdx.x;
  if (i < n) d[i] = (f16_t)s[i];
}

__global__ __launch_bounds__(256) void k_epilogue_va(
    const float* __restrict__ accum, const float* __restrict__ proj,
    f16_t* __restrict__ vaH)
{
  const int bs = blockIdx.x, b = bs >> 11, s = bs & 2047;
  for (int c = threadIdx.x; c < H_ * VF_; c += blockDim.x) {
    const int h = c >> 6, v = c & 63;
    const float val = accum[(((size_t)b * H_ + h) * S_ + s) * VF_ + v]
                    + 0.5f * proj[(size_t)bs * VF_ + v];
    vaH[(size_t)bs * (H_ * VF_) + c] = (f16_t)val;
  }
}

__global__ void k_fix1(f16_t* __restrict__ va)
{
  const int row = blockIdx.x * blockDim.x + threadIdx.x;
  if (row < B_ * S_) va[(size_t)row * KPAD_ + FEAT_] = (f16_t)1.f;
}

__global__ void k_final(const float* __restrict__ C, const float* __restrict__ masks,
                        float* __restrict__ out)
{
  const int bs = blockIdx.x;
  if (threadIdx.x < HS_)
    out[(size_t)bs * HS_ + threadIdx.x] = C[(size_t)bs * 64 + threadIdx.x] * masks[bs];
}

// ---------------------------------------------------------------------------
// Host orchestration
// ---------------------------------------------------------------------------
extern "C" void kernel_launch(void* const* d_in, const int* in_sizes, int n_in,
                              void* d_out, int out_size, void* d_ws, size_t ws_size,
                              hipStream_t stream)
{
  const float* latent = (const float*)d_in[0];
  const float* masks  = (const float*)d_in[1];
  const float* xemb   = (const float*)d_in[2];
  const float* ccls   = (const float*)d_in[4];
  const float* qlin   = (const float*)d_in[5];
  const float* klin   = (const float*)d_in[6];
  const float* vlin   = (const float*)d_in[7];
  const float* plin   = (const float*)d_in[8];
  const float* agg    = (const float*)d_in[9];
  const float* cfeat  = (const float*)d_in[10];
  float* out = (float*)d_out;

  char* w = (char*)d_ws;
  size_t off = 0;
  auto alloc = [&](size_t bytes) -> void* {
    void* p = w + off;
    off = (off + bytes + 255) & ~(size_t)255;
    return p;
  };

  const size_t BS = (size_t)B_ * S_;              // 16384
  f16_t* orig_h  = (f16_t*)alloc(BS * KPAD_ * 2);        // reused as va1_h
  f16_t* va1_h   = orig_h;                               // orig dead after layer-0 E GEMM
  f16_t* Wcat_h  = (f16_t*)alloc((size_t)KPAD_ * 1216 * 2);
  float* E       = (float*)alloc(BS * 1216 * 4);
  f16_t* eq_h    = (f16_t*)alloc((size_t)BH_ * S_ * DF_ * 2);
  f16_t* ekt_h   = (f16_t*)alloc((size_t)B_ * DF_ * S_ * 2);
  f16_t* ev_h    = (f16_t*)alloc((size_t)BH_ * S_ * VF_ * 2);
  float* proj0   = (float*)alloc(BS * VF_ * 4);
  float* proj1   = (float*)alloc(BS * VF_ * 4);
  f16_t* qland_h = (f16_t*)alloc((size_t)BH_ * NL_ * DF_ * 2);
  f16_t* klandt_h= (f16_t*)alloc((size_t)B_ * DF_ * NL_ * 2);
  float* possel  = (float*)alloc((size_t)H_ * NL_ * S_ * 4);
  float* possel2 = (float*)alloc((size_t)H_ * NL_ * NL_ * 4);
  float* big1    = (float*)alloc((size_t)BH_ * S_ * NL_ * 4);  // k1pre / scores / probs_h
  f16_t* probs_h = (f16_t*)big1;
  f16_t* k1_h    = (f16_t*)alloc((size_t)BH_ * S_ * NL_ * 2);
  float* k2pre   = (float*)alloc((size_t)BH_ * NL_ * NL_ * 4);
  float* key32   = (float*)alloc((size_t)BH_ * NL_ * NL_ * 4);
  f16_t* key_h   = (f16_t*)alloc((size_t)BH_ * NL_ * NL_ * 2);
  f16_t* value_h = (f16_t*)alloc((size_t)BH_ * NL_ * NL_ * 2);
  float* kvbuf   = (float*)alloc((size_t)BH_ * NL_ * NL_ * 4);
  f16_t* kv_h    = (f16_t*)alloc((size_t)BH_ * NL_ * NL_ * 2);
  f16_t* t_h     = (f16_t*)alloc((size_t)BH_ * NL_ * NL_ * 2);
  float* wbuf    = (float*)alloc((size_t)BH_ * NL_ * NL_ * 4);
  f16_t* w_h     = (f16_t*)alloc((size_t)BH_ * NL_ * NL_ * 2);
  float* ybuf    = (float*)alloc((size_t)BH_ * NL_ * NL_ * 4);
  f16_t* y_h     = (f16_t*)alloc((size_t)BH_ * NL_ * NL_ * 2);
  float* vnew    = (float*)alloc((size_t)BH_ * NL_ * NL_ * 4);
  f16_t* k3_h    = (f16_t*)alloc((size_t)BH_ * NL_ * S_ * 2);
  float* kv3     = (float*)alloc((size_t)BH_ * NL_ * VF_ * 4);
  f16_t* kv3_h   = (f16_t*)alloc((size_t)BH_ * NL_ * VF_ * 2);
  float* accum   = (float*)alloc((size_t)BH_ * S_ * VF_ * 4);
  f16_t* vaH_h   = (f16_t*)alloc(BS * H_ * VF_ * 2);
  f16_t* col0_h  = (f16_t*)alloc((size_t)512 * KPAD_ * 2);
  f16_t* colC_h  = (f16_t*)alloc((size_t)512 * 64 * 2);
  float* finalC  = (float*)alloc(BS * 64 * 4);
  float* gmax    = (float*)alloc(256);
  (void)ws_size; (void)in_sizes; (void)n_in; (void)out_size;

  // ---- one-time prep ----
  k_build_orig<<<dim3((unsigned)BS), dim3(64), 0, stream>>>(latent, masks, xemb, orig_h);
  k_prep_collect<<<dim3(512), dim3(256), 0, stream>>>(cfeat, ccls, col0_h, colC_h);

  const long NN = (long)NL_ * NL_;   // 16384, per-batch matrix stride

  for (int L = 0; L < 2; ++L) {
    const int Npad = (L == 0) ? 1216 : 1088;
    const f16_t* Ain = (L == 0) ? orig_h : va1_h;

    k_prep_w<<<dim3(KPAD_), dim3(256), 0, stream>>>(qlin, klin, vlin, plin, Wcat_h, L, Npad);
    k_possel<<<dim3(H_ * NL_), dim3(256), 0, stream>>>(agg, L, possel);
    k_possel2<<<dim3(H_ * NL_), dim3(128), 0, stream>>>(possel, possel2);

    // E = Ain(16384 x 1344) @ Wcat(1344 x Npad)  -- TDM weight-stationary GEMM
    gemm_tdm(stream, Ain, Wcat_h, E, false, (int)BS, Npad, KPAD_, KPAD_, Npad, Npad);

    k_landmark<<<dim3(B_ * NL_), dim3(64), 0, stream>>>(E, Npad, qland_h, klandt_h);
    k_extract<<<dim3((unsigned)BS), dim3(128), 0, stream>>>(E, Npad, L == 0 ? 1 : 0,
        eq_h, ekt_h, ev_h, proj0, proj1);

    // kernel_1 pre: Eq(S x DF) @ k_land^T(DF x NL), batched over (b,h)
    gemm(stream, eq_h, klandt_h, big1, false, S_, NL_, DF_, DF_, NL_, NL_,
         (long)S_ * DF_, 1, (long)DF_ * NL_, H_, (long)S_ * NL_, BH_);
    k_softmax1<<<dim3(BH_ * S_), dim3(128), 0, stream>>>(big1, possel, k1_h);

    // kernel_2 pre + softmax + global colsum max
    gemm(stream, qland_h, klandt_h, k2pre, false, NL_, NL_, DF_, DF_, NL_, NL_,
         (long)NL_ * DF_, 1, (long)DF_ * NL_, H_, NN, BH_);
    k_zero1<<<dim3(1), dim3(1), 0, stream>>>(gmax);
    k_softmax2<<<dim3(BH_), dim3(128), 0, stream>>>(k2pre, possel2, key32, key_h);
    k_colmax<<<dim3(BH_), dim3(128), 0, stream>>>(key32, gmax);
    k_initval<<<dim3(BH_ * NL_), dim3(128), 0, stream>>>(key32, gmax, value_h);

    // Newton-Schulz: value = 0.25*value@(13I - kv@(15I - kv@(7I - kv)))
    for (int it = 0; it < 6; ++it) {
      gemm(stream, key_h, value_h, kvbuf, false, NL_, NL_, NL_, NL_, NL_, NL_,
           NN, 1, NN, 1, NN, BH_);
      k_polyh<<<dim3(BH_ * NL_), dim3(128), 0, stream>>>(kvbuf, t_h, kv_h, 7.f, -1.f);
      gemm(stream, kv_h, t_h, wbuf, false, NL_, NL_, NL_, NL_, NL_, NL_,
           NN, 1, NN, 1, NN, BH_);
      k_polyh<<<dim3(BH_ * NL_), dim3(128), 0, stream>>>(wbuf, w_h, (f16_t*)nullptr, 15.f, -1.f);
      gemm(stream, kv_h, w_h, ybuf, false, NL_, NL_, NL_, NL_, NL_, NL_,
           NN, 1, NN, 1, NN, BH_);
      k_polyh<<<dim3(BH_ * NL_), dim3(128), 0, stream>>>(ybuf, y_h, (f16_t*)nullptr, 13.f, -1.f);
      gemm(stream, value_h, y_h, vnew, false, NL_, NL_, NL_, NL_, NL_, NL_,
           NN, 1, NN, 1, NN, BH_);
      k_polyh<<<dim3(BH_ * NL_), dim3(128), 0, stream>>>(vnew, value_h, (f16_t*)nullptr, 0.f, 0.25f);
    }

    // kernel_3: scores = q_land @ Ek^T, + mask + pos_sel, softmax over S
    gemm(stream, qland_h, ekt_h, big1, false, NL_, S_, DF_, DF_, S_, S_,
         (long)NL_ * DF_, 1, (long)DF_ * S_, H_, (long)NL_ * S_, BH_);
    k_softmax3<<<dim3(BH_ * NL_), dim3(256), 0, stream>>>(big1, possel, masks, k3_h);

    // kv3 = kernel_3 @ Ev
    gemm(stream, k3_h, ev_h, kv3, false, NL_, VF_, S_, S_, VF_, VF_,
         (long)NL_ * S_, 1, (long)S_ * VF_, 1, (long)NL_ * VF_, BH_);
    k_f32tof16<<<dim3((unsigned)(((long)BH_ * NL_ * VF_ + 255) / 256)), dim3(256), 0, stream>>>(
        kv3, kv3_h, (long)BH_ * NL_ * VF_);

    // probs = kernel_1 @ inv(kernel_2)   (f16 output, reuses big1)
    gemm(stream, k1_h, value_h, probs_h, true, S_, NL_, NL_, NL_, NL_, NL_,
         (long)S_ * NL_, 1, NN, 1, (long)S_ * NL_, BH_);
    // accum = probs @ kv3
    gemm(stream, probs_h, kv3_h, accum, false, S_, VF_, NL_, NL_, VF_, VF_,
         (long)S_ * NL_, 1, (long)NL_ * VF_, 1, (long)S_ * VF_, BH_);

    // va = accum + proj/2, heads interleaved -> (B,S,512) f16
    k_epilogue_va<<<dim3((unsigned)BS), dim3(256), 0, stream>>>(
        accum, L == 0 ? proj0 : proj1, vaH_h);

    if (L == 0) {
      // next-layer input: vaH @ collect_feats -> f16 (B,S,1344), then set 1-col
      gemm_tdm(stream, vaH_h, col0_h, va1_h, true, (int)BS, KPAD_, 512, 512, KPAD_, KPAD_);
      k_fix1<<<dim3((unsigned)((BS + 255) / 256)), dim3(256), 0, stream>>>(va1_h);
    } else {
      gemm_tdm(stream, vaH_h, colC_h, finalC, false, (int)BS, 64, 512, 512, 64, 64);
      k_final<<<dim3((unsigned)BS), dim3(64), 0, stream>>>(finalC, masks, out);
    }
  }
}
